// GNNEncoder_29910152249702
// MI455X (gfx1250) — matrix operand order repaired and checked by reading, not compiled, not run on hardware
//
#include <hip/hip_runtime.h>

typedef __attribute__((ext_vector_type(8)))  float  v8f;
typedef __attribute__((ext_vector_type(16))) __bf16 v16bf;
typedef __attribute__((ext_vector_type(8)))  __bf16 v8bf;

// ---------- helpers ----------
__device__ __forceinline__ unsigned short f2bf(float f) {
    unsigned int u = __float_as_uint(f);
    u += 0x7FFFu + ((u >> 16) & 1u);   // round-to-nearest-even
    return (unsigned short)(u >> 16);
}

// ---------- zero a float buffer (float4 granular) ----------
__global__ void zero_kernel(float4* __restrict__ p, int n4) {
    int i = blockIdx.x * blockDim.x + threadIdx.x;
    if (i < n4) p[i] = make_float4(0.f, 0.f, 0.f, 0.f);
}

// ---------- degree: deg[dst] += 1 ----------
__global__ void deg_kernel(const int* __restrict__ dst, float* __restrict__ deg, int E) {
    int e = blockIdx.x * blockDim.x + threadIdx.x;
    if (e < E) unsafeAtomicAdd(&deg[dst[e]], 1.0f);
}

// ---------- rdeg = 1/max(deg,1), in place ----------
__global__ void rdeg_kernel(float* __restrict__ deg, int N) {
    int i = blockIdx.x * blockDim.x + threadIdx.x;
    if (i < N) deg[i] = 1.0f / fmaxf(deg[i], 1.0f);
}

// ---------- pack 6 weight matrices (K,N) f32 -> (N,K) bf16 ----------
__global__ void wprep_kernel(const float* __restrict__ Wl0, const float* __restrict__ Wr0,
                             const float* __restrict__ Wl,  const float* __restrict__ Wr,
                             unsigned short* __restrict__ Wt) {
    int i = blockIdx.x * blockDim.x + threadIdx.x;      // 6 * 128 * 128
    if (i >= 6 * 16384) return;
    int m = i >> 14;
    int nk = i & 16383;
    int n = nk >> 7;
    int k = nk & 127;
    const float* W;
    switch (m) {
        case 0: W = Wl0;          break;
        case 1: W = Wr0;          break;
        case 2: W = Wl;           break;
        case 3: W = Wr;           break;
        case 4: W = Wl + 16384;   break;
        default: W = Wr + 16384;  break;
    }
    Wt[i] = f2bf(W[k * 128 + n]);   // transpose: Wt[n][k]
}

// ---------- scatter: one wave per edge, lane -> 4 channels ----------
__global__ void __launch_bounds__(256)
scatter_kernel(const float* __restrict__ h, const int* __restrict__ src,
               const int* __restrict__ dst, float* __restrict__ msg, int E) {
    int wid  = (blockIdx.x * blockDim.x + threadIdx.x) >> 5;
    int lane = threadIdx.x & 31;
    if (wid >= E) return;
    int s = src[wid];
    int d = dst[wid];
    float4 v = ((const float4*)(h + (size_t)s * 128))[lane];
    float* mp = msg + (size_t)d * 128 + lane * 4;
    unsafeAtomicAdd(mp + 0, v.x);
    unsafeAtomicAdd(mp + 1, v.y);
    unsafeAtomicAdd(mp + 2, v.z);
    unsafeAtomicAdd(mp + 3, v.w);
}

// ---------- mean+bf16 staging: meanbf = bf16(msg*rdeg), hbf = bf16(h) ----------
__global__ void convert_kernel(const float* __restrict__ msg, const float* __restrict__ rdeg,
                               const float* __restrict__ h,
                               unsigned short* __restrict__ meanbf,
                               unsigned short* __restrict__ hbf, int n4) {
    int i = blockIdx.x * blockDim.x + threadIdx.x;    // n4 = N*128/4 float4 packets
    if (i >= n4) return;
    int row = i >> 5;                                 // 32 float4 per 128-wide row
    float r = rdeg[row];
    float4 m  = ((const float4*)msg)[i];
    float4 hh = ((const float4*)h)[i];
    uint2 pm, ph;
    pm.x = (unsigned)f2bf(m.x * r) | ((unsigned)f2bf(m.y * r) << 16);
    pm.y = (unsigned)f2bf(m.z * r) | ((unsigned)f2bf(m.w * r) << 16);
    ph.x = (unsigned)f2bf(hh.x)    | ((unsigned)f2bf(hh.y)    << 16);
    ph.y = (unsigned)f2bf(hh.z)    | ((unsigned)f2bf(hh.w)    << 16);
    ((uint2*)meanbf)[i] = pm;
    ((uint2*)hbf)[i]    = ph;
}

// ---------- fused dual-GEMM + bias + LayerNorm + ReLU ----------
// One wave computes a 16x128 strip: out = mean@Wl + h@Wr + bl, then LN+ReLU.
__global__ void __launch_bounds__(256)
gemm_ln_relu_kernel(const unsigned short* __restrict__ meanbf,
                    const unsigned short* __restrict__ hbf,
                    const unsigned short* __restrict__ Wlt,   // bf16 [n][k]
                    const unsigned short* __restrict__ Wrt,   // bf16 [n][k]
                    const float* __restrict__ bias,
                    const float* __restrict__ gamma,
                    const float* __restrict__ beta,
                    float* __restrict__ out, int nstrips) {
    int wid = (blockIdx.x * blockDim.x + threadIdx.x) >> 5;
    if (wid >= nstrips) return;                // wave-uniform: EXEC stays all-ones
    int lane   = threadIdx.x & 31;
    int lane16 = lane & 15;
    int hi     = (lane >= 16) ? 1 : 0;

    v8f zero = {0.f, 0.f, 0.f, 0.f, 0.f, 0.f, 0.f, 0.f};
    v8f acc[8];
#pragma unroll
    for (int i = 0; i < 8; i++) acc[i] = zero;

    // A-fragment (16x32 bf16, contiguous k order, same as B): lane holds row lane16,
    // contiguous 16 K values {kb..kb+15}, kb shifted by +16 for upper half-wave.
    const __bf16* am_base = (const __bf16*)meanbf + ((size_t)(wid * 16 + lane16)) * 128 + hi * 16;
    const __bf16* ah_base = (const __bf16*)hbf    + ((size_t)(wid * 16 + lane16)) * 128 + hi * 16;

#pragma unroll
    for (int ks = 0; ks < 4; ks++) {
        v8bf am_lo = *(const v8bf*)(am_base + ks * 32);
        v8bf am_hi = *(const v8bf*)(am_base + ks * 32 + 8);
        v16bf a_mean = __builtin_shufflevector(am_lo, am_hi,
                        0, 1, 2, 3, 4, 5, 6, 7, 8, 9, 10, 11, 12, 13, 14, 15);
        v8bf ah_lo = *(const v8bf*)(ah_base + ks * 32);
        v8bf ah_hi = *(const v8bf*)(ah_base + ks * 32 + 8);
        v16bf a_h = __builtin_shufflevector(ah_lo, ah_hi,
                        0, 1, 2, 3, 4, 5, 6, 7, 8, 9, 10, 11, 12, 13, 14, 15);
#pragma unroll
        for (int nt = 0; nt < 8; nt++) {
            // B-fragment (32x16 bf16): lane holds column lane16, contiguous 16 K values
            size_t boff = ((size_t)(nt * 16 + lane16)) * 128 + ks * 32 + hi * 16;
            const __bf16* blp = (const __bf16*)Wlt + boff;
            v8bf bl0 = *(const v8bf*)(blp);
            v8bf bl1 = *(const v8bf*)(blp + 8);
            v16bf bfl = __builtin_shufflevector(bl0, bl1,
                        0, 1, 2, 3, 4, 5, 6, 7, 8, 9, 10, 11, 12, 13, 14, 15);
            acc[nt] = __builtin_amdgcn_wmma_f32_16x16x32_bf16(
                          false, a_mean, false, bfl, (short)0, acc[nt], false, false);
            const __bf16* brp = (const __bf16*)Wrt + boff;
            v8bf br0 = *(const v8bf*)(brp);
            v8bf br1 = *(const v8bf*)(brp + 8);
            v16bf bfr = __builtin_shufflevector(br0, br1,
                        0, 1, 2, 3, 4, 5, 6, 7, 8, 9, 10, 11, 12, 13, 14, 15);
            acc[nt] = __builtin_amdgcn_wmma_f32_16x16x32_bf16(
                          false, a_h, false, bfr, (short)0, acc[nt], false, false);
        }
    }

    // epilogue: bias + LayerNorm(128) + ReLU
    float bia[8], ga[8], bb[8];
#pragma unroll
    for (int nt = 0; nt < 8; nt++) {
        int c = nt * 16 + lane16;
        bia[nt] = bias[c];
        ga[nt]  = gamma[c];
        bb[nt]  = beta[c];
    }
    // C/D layout: lane<16 holds rows 0..7 (VGPR idx), lanes>=16 rows 8..15; col = nt*16+lane16
#pragma unroll
    for (int r = 0; r < 8; r++) {
        float v[8];
        float s1 = 0.f, s2 = 0.f;
#pragma unroll
        for (int nt = 0; nt < 8; nt++) {
            float t = acc[nt][r] + bia[nt];
            v[nt] = t;
            s1 += t;
            s2 += t * t;
        }
        // reduce over the 16 lanes that share this row (xor masks stay in half-wave)
#pragma unroll
        for (int m = 1; m < 16; m <<= 1) {
            s1 += __shfl_xor(s1, m, 32);
            s2 += __shfl_xor(s2, m, 32);
        }
        float mu   = s1 * (1.0f / 128.0f);
        float var  = s2 * (1.0f / 128.0f) - mu * mu;
        float rstd = rsqrtf(var + 1e-5f);
        int row = wid * 16 + (hi ? (8 + r) : r);
        float* op = out + (size_t)row * 128 + lane16;
#pragma unroll
        for (int nt = 0; nt < 8; nt++) {
            float y = (v[nt] - mu) * rstd * ga[nt] + bb[nt];
            op[nt * 16] = fmaxf(y, 0.0f);
        }
    }
}

// ---------------------------------------------------------------------------
extern "C" void kernel_launch(void* const* d_in, const int* in_sizes, int n_in,
                              void* d_out, int out_size, void* d_ws, size_t ws_size,
                              hipStream_t stream) {
    const float* x   = (const float*)d_in[0];
    const int*   ei  = (const int*)d_in[1];
    const float* Wl0 = (const float*)d_in[2];
    const float* bl0 = (const float*)d_in[3];
    const float* Wr0 = (const float*)d_in[4];
    const float* g0  = (const float*)d_in[5];
    const float* be0 = (const float*)d_in[6];
    const float* Wl  = (const float*)d_in[7];
    const float* bl  = (const float*)d_in[8];
    const float* Wr  = (const float*)d_in[9];
    const float* g   = (const float*)d_in[10];
    const float* be  = (const float*)d_in[11];

    const int N = in_sizes[0] / 128;
    const int E = in_sizes[1] / 2;
    const int* src = ei;
    const int* dst = ei + E;

    // workspace carve-up
    char* w = (char*)d_ws;
    float* msg = (float*)w;            w += (size_t)N * 128 * 4;
    float* hbuf = (float*)w;           w += (size_t)N * 128 * 4;
    unsigned short* meanbf = (unsigned short*)w; w += (size_t)N * 128 * 2;
    unsigned short* hbf    = (unsigned short*)w; w += (size_t)N * 128 * 2;
    float* rdeg = (float*)w;           w += (size_t)N * 4;
    unsigned short* Wt = (unsigned short*)w;     // 6 * 16384 bf16

    const int elems4 = N * 32;                 // N*128/4 float4 packets
    const int nstrips = N / 16;

    // ---- one-time prep: degree + packed weights ----
    zero_kernel<<<(N / 4 + 255) / 256, 256, 0, stream>>>((float4*)rdeg, N / 4);
    deg_kernel<<<(E + 255) / 256, 256, 0, stream>>>(dst, rdeg, E);
    rdeg_kernel<<<(N + 255) / 256, 256, 0, stream>>>(rdeg, N);
    wprep_kernel<<<(6 * 16384 + 255) / 256, 256, 0, stream>>>(Wl0, Wr0, Wl, Wr, Wt);

    const float* hcur = x;
    for (int l = 0; l < 3; l++) {
        const unsigned short* Wlt = Wt + (size_t)(2 * l) * 16384;
        const unsigned short* Wrt = Wt + (size_t)(2 * l + 1) * 16384;
        const float* bias  = (l == 0) ? bl0 : bl + (size_t)(l - 1) * 128;
        const float* gamma = (l == 0) ? g0  : g  + (size_t)(l - 1) * 128;
        const float* beta  = (l == 0) ? be0 : be + (size_t)(l - 1) * 128;
        float* hout = (l == 2) ? (float*)d_out : hbuf;

        zero_kernel<<<(elems4 + 255) / 256, 256, 0, stream>>>((float4*)msg, elems4);
        scatter_kernel<<<(E + 7) / 8, 256, 0, stream>>>(hcur, src, dst, msg, E);
        convert_kernel<<<(elems4 + 255) / 256, 256, 0, stream>>>(msg, rdeg, hcur,
                                                                 meanbf, hbf, elems4);
        gemm_ln_relu_kernel<<<(nstrips + 7) / 8, 256, 0, stream>>>(
            meanbf, hbf, Wlt, Wrt, bias, gamma, beta, hout, nstrips);

        hcur = hbuf;
    }
}
